// Attention_81372450390026
// MI455X (gfx1250) — matrix operand order repaired and checked by reading, not compiled, not run on hardware
//
#include <hip/hip_runtime.h>
#include <hip/hip_bf16.h>

#define HEADS_C 6
#define DHEAD 50
#define INNER_C 300
#define DIM_C 768
#define HW_C 100
#define EXTRA_C 150
#define POOL_C 10
#define NTOK_C (HW_C * HW_C + EXTRA_C)   // 10150
#define BATCH_C 8
#define NQ_C 100
#define NCHUNK_C 8
#define CHUNK_LEN ((NTOK_C + NCHUNK_C - 1) / NCHUNK_C)  // 1269
#define SCALE_C 0.14142135623730951f     // DIM_HEAD^-0.5

typedef __attribute__((ext_vector_type(16))) __bf16 v16bf;
typedef __attribute__((ext_vector_type(8)))  __bf16 v8bf;
typedef __attribute__((ext_vector_type(8)))  float  v8f;
typedef __attribute__((ext_vector_type(4)))  float  f32x4;
typedef __attribute__((ext_vector_type(4)))  unsigned int u32x4;
typedef __attribute__((ext_vector_type(2)))  unsigned int u32x2;

__device__ __forceinline__ unsigned short f32_to_bf16_rne(float f) {
  union { float f; unsigned int u; } cv; cv.f = f;
  unsigned int u = cv.u;
  u += 0x7FFFu + ((u >> 16) & 1u);
  return (unsigned short)(u >> 16);
}

__device__ __forceinline__ unsigned int pack_bf16x2(float lo, float hi) {
  return (unsigned int)f32_to_bf16_rne(lo) | ((unsigned int)f32_to_bf16_rne(hi) << 16);
}

// ---------------------------------------------------------------------------
// C[M,N] = A[M,K] @ W[N,K]^T (+bias). A fp32 or bf16(raw ushort); W fp32.
// bf16 WMMA, f32 accumulate. Block tile 128x64, 8 waves (4x2), 2x2 frags/wave.
// Staging is branchless: row/col indices are clamped (garbage rows/cols are
// never stored), and only the B-side K-tail is zero-filled (A garbage * 0 = 0).
// Requires K % 4 == 0.
// ---------------------------------------------------------------------------
template<bool A_BF16, bool ADD_BIAS>
__global__ __launch_bounds__(256)
void gemm_abT_wmma(const void* __restrict__ Aptr,
                   const float* __restrict__ Wptr,
                   const float* __restrict__ bias,
                   float* __restrict__ Cptr,
                   int M, int N, int K)
{
  constexpr int BM = 128, BN = 64, BK = 32;
  constexpr int LDA = BK + 8;  // pad to dodge bank conflicts
  __shared__ unsigned short As[BM][LDA];
  __shared__ unsigned short Bs[BN][LDA];

  const int tid  = threadIdx.x;
  const int lane = tid & 31;
  const int wave = tid >> 5;
  const int l15  = lane & 15;
  const int hi16 = lane >> 4;           // 0 or 1
  const int rbase = (wave & 3) * 32;    // wave M offset
  const int cbase = (wave >> 2) * 32;   // wave N offset

  const v8f zero8 = {0.f,0.f,0.f,0.f,0.f,0.f,0.f,0.f};
  v8f acc[2][2];
  #pragma unroll
  for (int i = 0; i < 2; ++i)
    #pragma unroll
    for (int j = 0; j < 2; ++j) acc[i][j] = zero8;

  const long Arow0 = (long)blockIdx.x * BM;
  const long Bcol0 = (long)blockIdx.y * BN;

  // ---- staging thread mapping (fixed across k-loop) ----
  const int arow_l = tid >> 1;            // 0..127
  const int aseg   = (tid & 1) * 16;      // 0 / 16
  long gm = Arow0 + arow_l; if (gm >= M) gm = M - 1;   // clamp: garbage rows OK
  const float*          arowF = (const float*)Aptr + gm * (long)K;
  const unsigned short* arowH = (const unsigned short*)Aptr + gm * (long)K;

  const int brow_l = tid >> 2;            // 0..63
  const int bseg   = (tid & 3) * 8;       // 0/8/16/24
  long gn = Bcol0 + brow_l; if (gn >= N) gn = N - 1;   // clamp: garbage cols OK
  const float* wrow = Wptr + gn * (long)K;

  for (int kb = 0; kb < K; kb += BK) {
    // ---- stage A tile: 128 rows x 32 k, bf16, vectorized & branchless ----
    if constexpr (A_BF16) {
      // 16 shorts = 4x 8-byte loads (rows only 8B aligned). Tail reads rely
      // on a 256B pad after the bf16 buffer; garbage is nulled by B's zeros.
      u32x2 u[4];
      #pragma unroll
      for (int i = 0; i < 4; ++i)
        u[i] = *(const u32x2*)(arowH + kb + aseg + 4 * i);
      u32x4 s0 = {u[0].x, u[0].y, u[1].x, u[1].y};
      u32x4 s1 = {u[2].x, u[2].y, u[3].x, u[3].y};
      *(u32x4*)&As[arow_l][aseg]     = s0;
      *(u32x4*)&As[arow_l][aseg + 8] = s1;
    } else {
      f32x4 v[4];
      #pragma unroll
      for (int i = 0; i < 4; ++i) {
        int k4 = kb + aseg + 4 * i;
        if (k4 > K - 4) k4 = K - 4;     // clamp (never triggers when K%32==0)
        v[i] = *(const f32x4*)(arowF + k4);
      }
      u32x4 s0 = {pack_bf16x2(v[0].x, v[0].y), pack_bf16x2(v[0].z, v[0].w),
                  pack_bf16x2(v[1].x, v[1].y), pack_bf16x2(v[1].z, v[1].w)};
      u32x4 s1 = {pack_bf16x2(v[2].x, v[2].y), pack_bf16x2(v[2].z, v[2].w),
                  pack_bf16x2(v[3].x, v[3].y), pack_bf16x2(v[3].z, v[3].w)};
      *(u32x4*)&As[arow_l][aseg]     = s0;
      *(u32x4*)&As[arow_l][aseg + 8] = s1;
    }
    // ---- stage B tile: 64 cols x 32 k; zero-fill K tail (branchless) ----
    {
      f32x4 v[2];
      #pragma unroll
      for (int i = 0; i < 2; ++i) {
        int k4 = kb + bseg + 4 * i;
        int k4c = k4 > K - 4 ? K - 4 : k4;
        f32x4 t = *(const f32x4*)(wrow + k4c);
        if (k4 >= K) t = (f32x4){0.f, 0.f, 0.f, 0.f};  // K%4==0 -> whole-vec select
        v[i] = t;
      }
      u32x4 s = {pack_bf16x2(v[0].x, v[0].y), pack_bf16x2(v[0].z, v[0].w),
                 pack_bf16x2(v[1].x, v[1].y), pack_bf16x2(v[1].z, v[1].w)};
      *(u32x4*)&Bs[brow_l][bseg] = s;
    }
    // prefetch next A k-tile into cache
    if (kb + BK < K) {
      if constexpr (A_BF16) __builtin_prefetch(arowH + kb + BK + aseg, 0, 0);
      else                  __builtin_prefetch(arowF + kb + BK + aseg, 0, 0);
    }
    __syncthreads();

    // A frag layout (16x32 bf16): lane<16 -> row=lane, K 0..7 & 16..23;
    //                             lane>=16 -> row=lane-16, K 8..15 & 24..31.
    v16bf afr[2], bfr[2];
    #pragma unroll
    for (int mi = 0; mi < 2; ++mi) {
      int row = rbase + mi * 16 + l15;
      int k0  = hi16 * 8;
      v8bf lo = *(const v8bf*)&As[row][k0];
      v8bf hi = *(const v8bf*)&As[row][k0 + 16];
      afr[mi] = __builtin_shufflevector(lo, hi, 0,1,2,3,4,5,6,7,8,9,10,11,12,13,14,15);
    }
    // B frag layout (32x16 bf16): lane<16 -> col=lane, K 0..7 & 16..23;
    //                             lane>=16 -> col=lane-16, K 8..15 & 24..31.
    #pragma unroll
    for (int ni = 0; ni < 2; ++ni) {
      int col = cbase + ni * 16 + l15;
      int k0  = hi16 * 8;
      v8bf lo = *(const v8bf*)&Bs[col][k0];
      v8bf hi = *(const v8bf*)&Bs[col][k0 + 16];
      bfr[ni] = __builtin_shufflevector(lo, hi, 0,1,2,3,4,5,6,7,8,9,10,11,12,13,14,15);
    }
    #pragma unroll
    for (int mi = 0; mi < 2; ++mi)
      #pragma unroll
      for (int ni = 0; ni < 2; ++ni)
        acc[mi][ni] = __builtin_amdgcn_wmma_f32_16x16x32_bf16(
            false, afr[mi], false, bfr[ni], (short)0, acc[mi][ni], false, false);

    __syncthreads();
  }

  // C/D layout: VGPR r -> (M = r + 8*hi16, N = l15) within the 16x16 frag
  #pragma unroll
  for (int mi = 0; mi < 2; ++mi)
    #pragma unroll
    for (int ni = 0; ni < 2; ++ni)
      #pragma unroll
      for (int r = 0; r < 8; ++r) {
        long row = Arow0 + rbase + mi * 16 + r + hi16 * 8;
        int  col = (int)Bcol0 + cbase + ni * 16 + l15;
        if (row < M && col < N) {
          float v = acc[mi][ni][r];
          if constexpr (ADD_BIAS) v += bias[col];
          Cptr[row * (long)N + col] = v;
        }
      }
}

// ---------------------------------------------------------------------------
// Adaptive avg-pool 100x100 -> 10x10 of spatial tokens; output [b,h,q,d]
// ---------------------------------------------------------------------------
__global__ void pool_kernel(const float* __restrict__ w, float* __restrict__ rep)
{
  int idx = blockIdx.x * blockDim.x + threadIdx.x;
  if (idx >= BATCH_C * NQ_C * INNER_C) return;
  int c = idx % INNER_C;
  int q = (idx / INNER_C) % NQ_C;
  int b = idx / (INNER_C * NQ_C);
  int py = q / POOL_C, px = q % POOL_C;
  float s = 0.f;
  for (int iy = 0; iy < 10; ++iy)
    for (int ix = 0; ix < 10; ++ix) {
      int n = (py * 10 + iy) * HW_C + (px * 10 + ix);
      s += w[((size_t)b * NTOK_C + n) * INNER_C + c];
    }
  s *= 0.01f;
  int h = c / DHEAD, d = c % DHEAD;
  rep[(((size_t)b * HEADS_C + h) * NQ_C + q) * DHEAD + d] = s;
}

// ---------------------------------------------------------------------------
// Stage 1 flash attention (rep attends to all tokens), chunked over N.
// Emits per-(bh,chunk): row max m, row sumexp z, unnormalized O.
// ---------------------------------------------------------------------------
__global__ __launch_bounds__(256)
void stage1_partial(const float* __restrict__ w, const float* __restrict__ rep,
                    float* __restrict__ pm, float* __restrict__ pz,
                    float* __restrict__ pO)
{
  const int bh = blockIdx.x, chunk = blockIdx.y;
  const int b = bh / HEADS_C, h = bh % HEADS_C;
  const int tid = threadIdx.x;

  __shared__ float Qs[NQ_C * DHEAD];
  __shared__ float Ks[64][DHEAD + 3];
  __shared__ float Ss[NQ_C][65];
  __shared__ float mrow[NQ_C], zrow[NQ_C], rfac[NQ_C];

  const float* qsrc = rep + (size_t)bh * NQ_C * DHEAD;
  for (int i = tid; i < NQ_C * DHEAD; i += 256) Qs[i] = qsrc[i];
  if (tid < NQ_C) { mrow[tid] = -1e30f; zrow[tid] = 0.f; }

  float accO[20];
  #pragma unroll
  for (int i = 0; i < 20; ++i) accO[i] = 0.f;

  const int n0 = chunk * CHUNK_LEN;
  const int n1 = min(NTOK_C, n0 + CHUNK_LEN);
  __syncthreads();

  for (int nt = n0; nt < n1; nt += 64) {
    const int tn = min(64, n1 - nt);
    for (int i = tid; i < tn * DHEAD; i += 256) {
      int j = i / DHEAD, d = i - j * DHEAD;
      Ks[j][d] = w[((size_t)b * NTOK_C + (nt + j)) * INNER_C + h * DHEAD + d];
    }
    __syncthreads();

    for (int i = tid; i < NQ_C * 64; i += 256) {
      int q = i >> 6, j = i & 63;
      float s = -1e30f;
      if (j < tn) {
        s = 0.f;
        const float* qp = &Qs[q * DHEAD];
        #pragma unroll 10
        for (int d = 0; d < DHEAD; ++d) s += qp[d] * Ks[j][d];
        s *= SCALE_C;
      }
      Ss[q][j] = s;
    }
    __syncthreads();

    if (tid < NQ_C) {            // online softmax row update
      int q = tid;
      float tmax = -1e30f;
      for (int j = 0; j < 64; ++j) tmax = fmaxf(tmax, Ss[q][j]);
      float mold = mrow[q];
      float mnew = fmaxf(mold, tmax);
      float r = __expf(mold - mnew);
      float ps = 0.f;
      for (int j = 0; j < 64; ++j) {
        float p = __expf(Ss[q][j] - mnew);
        Ss[q][j] = p; ps += p;
      }
      zrow[q] = zrow[q] * r + ps;
      mrow[q] = mnew;
      rfac[q] = r;
    }
    __syncthreads();

    for (int i = 0, p = tid; p < NQ_C * DHEAD; ++i, p += 256) {
      int q = p / DHEAD, d = p - q * DHEAD;
      float a = accO[i] * rfac[q];
      #pragma unroll 16
      for (int j = 0; j < 64; ++j) a += Ss[q][j] * Ks[j][d];  // stale K rows hit P==0
      accO[i] = a;
    }
    __syncthreads();
  }

  size_t pb = ((size_t)bh * NCHUNK_C + chunk) * NQ_C;
  if (tid < NQ_C) { pm[pb + tid] = mrow[tid]; pz[pb + tid] = zrow[tid]; }
  size_t ob = ((size_t)bh * NCHUNK_C + chunk) * (NQ_C * DHEAD);
  for (int i = 0, p = tid; p < NQ_C * DHEAD; ++i, p += 256) pO[ob + p] = accO[i];
}

__global__ __launch_bounds__(128)
void stage1_combine(const float* __restrict__ pm, const float* __restrict__ pz,
                    const float* __restrict__ pO, const float* __restrict__ rep,
                    const float* __restrict__ step_rep,
                    float* __restrict__ mglob, float* __restrict__ zglob,
                    float* __restrict__ rep_new)
{
  const int bh = blockIdx.x;
  const int h = bh % HEADS_C;
  const int tid = threadIdx.x;
  __shared__ float zinv[NQ_C];
  __shared__ float coef[NCHUNK_C][NQ_C];
  if (tid < NQ_C) {
    int q = tid;
    size_t base = (size_t)bh * NCHUNK_C * NQ_C;
    float m = -1e30f;
    for (int i = 0; i < NCHUNK_C; ++i) m = fmaxf(m, pm[base + i * NQ_C + q]);
    float z = 0.f;
    for (int i = 0; i < NCHUNK_C; ++i) {
      float c = __expf(pm[base + i * NQ_C + q] - m);
      coef[i][q] = c;
      z += pz[base + i * NQ_C + q] * c;
    }
    mglob[(size_t)bh * NQ_C + q] = m;
    zglob[(size_t)bh * NQ_C + q] = z;
    zinv[q] = 1.f / z;
  }
  __syncthreads();
  const float sr = step_rep[h];
  size_t ob = (size_t)bh * NCHUNK_C * NQ_C * DHEAD;
  size_t rb = (size_t)bh * NQ_C * DHEAD;
  for (int p = tid; p < NQ_C * DHEAD; p += 128) {
    int q = p / DHEAD;
    float o = 0.f;
    for (int i = 0; i < NCHUNK_C; ++i)
      o += coef[i][q] * pO[ob + (size_t)i * NQ_C * DHEAD + p];
    rep_new[rb + p] = rep[rb + p] + sr * (o * zinv[q]);
  }
}

// ---------------------------------------------------------------------------
// Stage 2: rep self-attention, 100x100, fully in LDS. xd2 = step_x * attn2 @ R
// ---------------------------------------------------------------------------
__global__ __launch_bounds__(256)
void stage2_kernel(const float* __restrict__ rep_new, const float* __restrict__ step_x,
                   float* __restrict__ xd2)
{
  const int bh = blockIdx.x;
  const int h = bh % HEADS_C;
  const int tid = threadIdx.x;
  __shared__ float R[NQ_C][DHEAD + 3];
  __shared__ float S2[NQ_C][NQ_C + 1];
  const float* src = rep_new + (size_t)bh * NQ_C * DHEAD;
  for (int i = tid; i < NQ_C * DHEAD; i += 256) {
    int q = i / DHEAD, d = i - q * DHEAD;
    R[q][d] = src[i];
  }
  __syncthreads();
  for (int i = tid; i < NQ_C * NQ_C; i += 256) {
    int q = i / NQ_C, k = i - q * NQ_C;
    float s = 0.f;
    #pragma unroll 10
    for (int d = 0; d < DHEAD; ++d) s += R[q][d] * R[k][d];
    S2[q][k] = s * SCALE_C;
  }
  __syncthreads();
  if (tid < NQ_C) {
    int q = tid;
    float m = -1e30f;
    for (int k = 0; k < NQ_C; ++k) m = fmaxf(m, S2[q][k]);
    float z = 0.f;
    for (int k = 0; k < NQ_C; ++k) { float p = __expf(S2[q][k] - m); S2[q][k] = p; z += p; }
    float zi = 1.f / z;
    for (int k = 0; k < NQ_C; ++k) S2[q][k] *= zi;
  }
  __syncthreads();
  const float sx = step_x[h];
  float* dst = xd2 + (size_t)bh * NQ_C * DHEAD;
  for (int p = tid; p < NQ_C * DHEAD; p += 256) {
    int q = p / DHEAD, d = p - q * DHEAD;
    float a = 0.f;
    #pragma unroll 10
    for (int k = 0; k < NQ_C; ++k) a += S2[q][k] * R[k][d];
    dst[p] = sx * a;
  }
}

// ---------------------------------------------------------------------------
// Scatter: x_delta[b,n,h*50+d] = sum_q attn[q,n] * xd2[q,d]; attn recomputed
// from (Q=rep_pre, K=w, m, z). Output bf16 for the final WMMA GEMM.
// ---------------------------------------------------------------------------
__global__ __launch_bounds__(256)
void scatter_kernel(const float* __restrict__ w, const float* __restrict__ rep,
                    const float* __restrict__ mglob, const float* __restrict__ zglob,
                    const float* __restrict__ xd2, unsigned short* __restrict__ xdf)
{
  const int bh = blockIdx.y;
  const int b = bh / HEADS_C, h = bh % HEADS_C;
  const int tid = threadIdx.x;
  __shared__ float Qs[NQ_C][DHEAD];
  __shared__ float V2[NQ_C][DHEAD];
  __shared__ float msh[NQ_C], zish[NQ_C];
  const float* qsrc = rep + (size_t)bh * NQ_C * DHEAD;
  const float* vsrc = xd2 + (size_t)bh * NQ_C * DHEAD;
  for (int i = tid; i < NQ_C * DHEAD; i += 256) {
    ((float*)Qs)[i] = qsrc[i];
    ((float*)V2)[i] = vsrc[i];
  }
  if (tid < NQ_C) {
    msh[tid]  = mglob[(size_t)bh * NQ_C + tid];
    zish[tid] = 1.f / zglob[(size_t)bh * NQ_C + tid];
  }
  __syncthreads();

  const int n = blockIdx.x * 256 + tid;
  if (n >= NTOK_C) return;

  float kreg[DHEAD], acc[DHEAD];
  const float* wp = w + ((size_t)b * NTOK_C + n) * INNER_C + h * DHEAD;
  #pragma unroll 10
  for (int d = 0; d < DHEAD; ++d) { kreg[d] = wp[d]; acc[d] = 0.f; }

  for (int q = 0; q < NQ_C; ++q) {
    float s = 0.f;
    #pragma unroll 10
    for (int d = 0; d < DHEAD; ++d) s += Qs[q][d] * kreg[d];
    float p = __expf(s * SCALE_C - msh[q]) * zish[q];
    #pragma unroll 10
    for (int d = 0; d < DHEAD; ++d) acc[d] += p * V2[q][d];
  }
  unsigned short* op = xdf + ((size_t)b * NTOK_C + n) * INNER_C + h * DHEAD;
  #pragma unroll 10
  for (int d = 0; d < DHEAD; ++d) op[d] = f32_to_bf16_rne(acc[d]);
}

// ---------------------------------------------------------------------------
extern "C" void kernel_launch(void* const* d_in, const int* in_sizes, int n_in,
                              void* d_out, int out_size, void* d_ws, size_t ws_size,
                              hipStream_t stream)
{
  (void)in_sizes; (void)n_in; (void)out_size; (void)ws_size;
  const float* x        = (const float*)d_in[0];
  const float* proj_w   = (const float*)d_in[1];
  const float* step_x   = (const float*)d_in[2];
  const float* step_rep = (const float*)d_in[3];
  const float* out_w    = (const float*)d_in[4];
  const float* out_b    = (const float*)d_in[5];
  float* out = (float*)d_out;

  const int M = BATCH_C * NTOK_C;  // 81200
  char* ws = (char*)d_ws;
  size_t off = 0;
  auto take = [&](size_t bytes) -> char* {
    char* p = ws + off;
    off += (bytes + 255) & ~(size_t)255;
    return p;
  };
  float* w        = (float*)take((size_t)M * INNER_C * 4);                                // 97.4 MB
  float* rep_pre  = (float*)take((size_t)BATCH_C * HEADS_C * NQ_C * DHEAD * 4);
  float* rep_new  = (float*)take((size_t)BATCH_C * HEADS_C * NQ_C * DHEAD * 4);
  float* xd2      = (float*)take((size_t)BATCH_C * HEADS_C * NQ_C * DHEAD * 4);
  float* pm       = (float*)take((size_t)BATCH_C * HEADS_C * NCHUNK_C * NQ_C * 4);
  float* pz       = (float*)take((size_t)BATCH_C * HEADS_C * NCHUNK_C * NQ_C * 4);
  float* pO       = (float*)take((size_t)BATCH_C * HEADS_C * NCHUNK_C * NQ_C * DHEAD * 4); // 7.7 MB
  float* mglob    = (float*)take((size_t)BATCH_C * HEADS_C * NQ_C * 4);
  float* zglob    = (float*)take((size_t)BATCH_C * HEADS_C * NQ_C * 4);
  unsigned short* xdf = (unsigned short*)take((size_t)M * INNER_C * 2 + 256);              // 48.7 MB (+tail pad)

  // 1) w = x @ proj_w^T   (bf16 WMMA, f32 accumulate)
  {
    dim3 grid((M + 127) / 128, (INNER_C + 63) / 64);
    gemm_abT_wmma<false, false><<<grid, 256, 0, stream>>>(x, proj_w, nullptr, w,
                                                          M, INNER_C, DIM_C);
  }
  // 2) adaptive avg pool -> rep_pre [b,h,q,d]
  {
    int total = BATCH_C * NQ_C * INNER_C;
    pool_kernel<<<(total + 255) / 256, 256, 0, stream>>>(w, rep_pre);
  }
  // 3) stage-1 flash attention (partials + combine)
  {
    dim3 grid(BATCH_C * HEADS_C, NCHUNK_C);
    stage1_partial<<<grid, 256, 0, stream>>>(w, rep_pre, pm, pz, pO);
    stage1_combine<<<BATCH_C * HEADS_C, 128, 0, stream>>>(pm, pz, pO, rep_pre, step_rep,
                                                          mglob, zglob, rep_new);
  }
  // 4) stage-2 rep self-attention
  stage2_kernel<<<BATCH_C * HEADS_C, 256, 0, stream>>>(rep_new, step_x, xd2);
  // 5) scatter back through attn^T (attn recomputed from m/z) -> bf16 x_delta
  {
    dim3 grid((NTOK_C + 255) / 256, BATCH_C * HEADS_C);
    scatter_kernel<<<grid, 256, 0, stream>>>(w, rep_pre, mglob, zglob, xd2, xdf);
  }
  // 6) out = x_delta @ out_w^T + out_b   (bf16 WMMA, f32 accumulate)
  {
    dim3 grid((M + 127) / 128, (DIM_C + 63) / 64);
    gemm_abT_wmma<true, true><<<grid, 256, 0, stream>>>(xdf, out_w, out_b, out,
                                                        M, DIM_C, INNER_C);
  }
}
